// group_query_attn_20598663152352
// MI455X (gfx1250) — matrix operand order repaired and checked by reading, not compiled, not run on hardware
//
#include <hip/hip_runtime.h>
#include <hip/hip_bf16.h>

typedef __bf16 bf16;
typedef __attribute__((ext_vector_type(16))) __bf16 v16bf;
typedef __attribute__((ext_vector_type(8)))  __bf16 v8bf;
typedef __attribute__((ext_vector_type(8)))  float  v8f;
typedef __attribute__((ext_vector_type(4)))  float  v4f;

#define DIM      4096
#define Q_HEADS  32
#define KV_HEADS 8
#define HD       128
#define SLEN     2048
#define BATCH    2
#define TOKENS   (BATCH * SLEN)      // 4096

// ---------------------------------------------------------------------------
// CDNA5 bf16 fragment helpers (wave32 VGPR layouts, CDNA5 ISA)
//  A (16x32): lane l holds row M=l&15; K = kbase + {0..7} and kbase+8+{0..7},
//             kbase = (l>>4)*16  -> two 16B LDS loads.
//  B (32x16): lane l holds col N=l&15; K = (l>>4)*16 + {0..15} contiguous
//             -> one 32B LDS load when stored K-contiguous per column.
//  C/D (16x16 f32): lane l col N=l&15, VGPR r holds row M = r + (l>>4)*8.
// ---------------------------------------------------------------------------
__device__ __forceinline__ v16bf ldsA_frag(const bf16* base, int row, int kbase, int ldr) {
    const v8bf lo = *(const v8bf*)(base + row * ldr + kbase);
    const v8bf hi = *(const v8bf*)(base + row * ldr + kbase + 8);
    v16bf r;
#pragma unroll
    for (int i = 0; i < 8; ++i) { r[i] = lo[i]; r[8 + i] = hi[i]; }
    return r;
}

__device__ __forceinline__ v16bf ldsB_frag(const bf16* base, int col, int kbase, int ldr) {
    return *(const v16bf*)(base + col * ldr + kbase);   // 32B aligned by construction
}

__device__ __forceinline__ v8f wmma_bf16(v16bf a, v16bf b, v8f c) {
    return __builtin_amdgcn_wmma_f32_16x16x32_bf16(false, a, false, b, (short)0, c,
                                                   false, false);
}

// ---------------------------------------------------------------------------
// GEMM: C[M,N] = A[M,K] * B[K,N] + bias  (bf16 WMMA, fp32 accum)
//  A: fp32 (converted to bf16 while staging) or bf16 (A_BF16)
//  B: fp32, converted to bf16 while staging (stored transposed [N][K] in LDS)
//  C: bf16 (workspace) or fp32 (final output)
// Block: 256 threads = 8 waves; tile 128x128, BK=32; wave = 32x64 (2x4 frags).
// ---------------------------------------------------------------------------
template <bool A_BF16, bool OUT_BF16>
__global__ __launch_bounds__(256)
void gemm_bias_wmma(const void* __restrict__ Ap, const float* __restrict__ B,
                    const float* __restrict__ bias, void* __restrict__ Cp,
                    int M, int N, int K) {
    __shared__ __align__(32) bf16 ldsA[128 * 32];   // [M][K]
    __shared__ __align__(32) bf16 ldsB[128 * 32];   // [N][K] (transposed)

    const int tid  = threadIdx.x;
    const int lane = tid & 31;
    const int wave = tid >> 5;
    const int l15  = lane & 15;
    const int lh   = lane >> 4;
    const int wm   = wave >> 1;           // 0..3 -> 32-row strip
    const int wn   = wave & 1;            // 0..1 -> 64-col strip
    const int m0   = blockIdx.y * 128;
    const int n0   = blockIdx.x * 128;

    const float* Af = (const float*)Ap;
    const bf16*  Ab = (const bf16*)Ap;

    v8f acc[2][4];
#pragma unroll
    for (int mi = 0; mi < 2; ++mi)
#pragma unroll
        for (int ni = 0; ni < 4; ++ni)
#pragma unroll
            for (int e = 0; e < 8; ++e) acc[mi][ni][e] = 0.f;

    const int ktiles = K >> 5;
    for (int kt = 0; kt < ktiles; ++kt) {
        const int kk = kt * 32;
        __syncthreads();
        // ---- stage A tile (128x32) ----
        if (A_BF16) {
#pragma unroll
            for (int i = 0; i < 2; ++i) {
                int task = tid + i * 256;           // 128 rows * 4 chunks
                int r = task >> 2, c = (task & 3) * 8;
                *(v8bf*)&ldsA[r * 32 + c] =
                    *(const v8bf*)(Ab + (size_t)(m0 + r) * K + kk + c);
            }
        } else {
#pragma unroll
            for (int i = 0; i < 4; ++i) {
                int task = tid + i * 256;           // 128 rows * 8 chunks
                int r = task >> 3, c = (task & 7) * 4;
                v4f va = *(const v4f*)(Af + (size_t)(m0 + r) * K + kk + c);
#pragma unroll
                for (int j = 0; j < 4; ++j) ldsA[r * 32 + c + j] = (bf16)va[j];
            }
        }
        // ---- stage B tile (32x128 -> LDS [N][K]) ----
#pragma unroll
        for (int i = 0; i < 4; ++i) {
            int task = tid + i * 256;               // 32 k * 32 n-chunks
            int k = task >> 5, n = (task & 31) * 4;
            v4f vb = *(const v4f*)(B + (size_t)(kk + k) * N + n0 + n);
#pragma unroll
            for (int j = 0; j < 4; ++j) ldsB[(n + j) * 32 + k] = (bf16)vb[j];
        }
        // prefetch next K-tile rows into cache (global_prefetch_b8)
        if (kt + 1 < ktiles) {
            __builtin_prefetch(B + (size_t)(kk + 32 + (tid >> 3)) * N + n0 + (tid & 7) * 16, 0, 1);
            if (A_BF16)
                __builtin_prefetch(Ab + (size_t)(m0 + (tid >> 1)) * K + kk + 32, 0, 1);
            else
                __builtin_prefetch(Af + (size_t)(m0 + (tid >> 1)) * K + kk + 32, 0, 1);
        }
        __syncthreads();

        v16bf af[2];
#pragma unroll
        for (int mi = 0; mi < 2; ++mi)
            af[mi] = ldsA_frag(ldsA, wm * 32 + mi * 16 + l15, lh * 16, 32);
#pragma unroll
        for (int ni = 0; ni < 4; ++ni) {
            v16bf bfr = ldsB_frag(ldsB, wn * 64 + ni * 16 + l15, lh * 16, 32);
#pragma unroll
            for (int mi = 0; mi < 2; ++mi)
                acc[mi][ni] = wmma_bf16(af[mi], bfr, acc[mi][ni]);
        }
    }

    // ---- epilogue: bias + store ----
#pragma unroll
    for (int mi = 0; mi < 2; ++mi)
#pragma unroll
        for (int ni = 0; ni < 4; ++ni) {
            int col = n0 + wn * 64 + ni * 16 + l15;
            float bv = bias[col];
#pragma unroll
            for (int r = 0; r < 8; ++r) {
                int row = m0 + wm * 32 + mi * 16 + lh * 8 + r;
                float val = acc[mi][ni][r] + bv;
                if (OUT_BF16)
                    ((bf16*)Cp)[(size_t)row * N + col] = (bf16)val;
                else
                    ((float*)Cp)[(size_t)row * N + col] = val;
            }
        }
}

// ---------------------------------------------------------------------------
// Flash-style GQA attention. One block = (batch b, q-head qh, 128-query tile).
// Q fragments held in registers; per 128-key tile: S = Q*K^T via WMMA,
// online softmax (shfl_xor reductions inside 16-lane halves), P re-staged
// through LDS into A-fragment layout, O += P*V via WMMA.
// LDS: lds_kp (K tile, then P tile) 32KB + lds_v 32KB = 64KB.
// ---------------------------------------------------------------------------
__global__ __launch_bounds__(256)
void gqa_attention_kernel(const bf16* __restrict__ Q, const bf16* __restrict__ Kt,
                          const bf16* __restrict__ Vt, bf16* __restrict__ Oa) {
    __shared__ __align__(32) bf16 lds_kp[128 * 128];  // K tile [sk][d] / P tile [q][sk]
    __shared__ __align__(32) bf16 lds_v [128 * 128];  // V tile transposed [d][sk]

    const int tid  = threadIdx.x;
    const int lane = tid & 31;
    const int wave = tid >> 5;          // 8 waves * 16 query rows
    const int l15  = lane & 15;
    const int lh   = lane >> 4;

    const int qtile = blockIdx.x & 15;
    const int qh    = (blockIdx.x >> 4) & 31;
    const int b     = blockIdx.x >> 9;
    const int kvh   = qh >> 2;

    const int    qrow0 = b * SLEN + qtile * 128;
    const bf16*  qbase = Q  + (size_t)qrow0 * DIM + qh * HD;
    const bf16*  kbase = Kt + (size_t)(b * SLEN) * (KV_HEADS * HD) + kvh * HD;
    const bf16*  vbase = Vt + (size_t)(b * SLEN) * (KV_HEADS * HD) + kvh * HD;

    // ---- stage Q tile, pin A-fragments in registers ----
#pragma unroll
    for (int i = 0; i < 8; ++i) {
        int task = tid + i * 256;                    // 128 rows * 16 chunks of 8
        int r = task >> 4, c = (task & 15) * 8;
        *(v8bf*)&lds_kp[r * 128 + c] = *(const v8bf*)(qbase + (size_t)r * DIM + c);
    }
    __syncthreads();
    v16bf qfrag[4];
#pragma unroll
    for (int kf = 0; kf < 4; ++kf)
        qfrag[kf] = ldsA_frag(lds_kp, wave * 16 + l15, kf * 32 + lh * 16, 128);

    v8f   o[8];
    float mrow[8], lrow[8];
#pragma unroll
    for (int nf = 0; nf < 8; ++nf)
#pragma unroll
        for (int e = 0; e < 8; ++e) o[nf][e] = 0.f;
#pragma unroll
    for (int r = 0; r < 8; ++r) { mrow[r] = -1e30f; lrow[r] = 0.f; }

    const float scale = 0.08838834764831845f;        // 1/sqrt(128)

    for (int kt = 0; kt < SLEN / 128; ++kt) {
        __syncthreads();                             // done with lds_kp/lds_v
        // ---- stage K tile [sk][d] and V tile transposed [d][sk] ----
#pragma unroll
        for (int i = 0; i < 8; ++i) {
            int task = tid + i * 256;
            int r = task >> 4, c = (task & 15) * 8;
            const size_t grow = (size_t)(kt * 128 + r) * (KV_HEADS * HD);
            *(v8bf*)&lds_kp[r * 128 + c] = *(const v8bf*)(kbase + grow + c);
            v8bf vv = *(const v8bf*)(vbase + grow + c);
#pragma unroll
            for (int j = 0; j < 8; ++j) lds_v[(c + j) * 128 + r] = vv[j];
        }
        __syncthreads();

        // ---- S = Q * K^T (16 x 128 per wave) ----
        v8f s[8];
#pragma unroll
        for (int nf = 0; nf < 8; ++nf)
#pragma unroll
            for (int e = 0; e < 8; ++e) s[nf][e] = 0.f;
#pragma unroll
        for (int kf = 0; kf < 4; ++kf)
#pragma unroll
            for (int nf = 0; nf < 8; ++nf) {
                v16bf bk = ldsB_frag(lds_kp, nf * 16 + l15, kf * 32 + lh * 16, 128);
                s[nf] = wmma_bf16(qfrag[kf], bk, s[nf]);
            }

        // ---- online softmax (rows live in same 16-lane half) ----
        float tmax[8], rs[8], scv[8];
#pragma unroll
        for (int r = 0; r < 8; ++r) tmax[r] = -1e30f;
#pragma unroll
        for (int nf = 0; nf < 8; ++nf)
#pragma unroll
            for (int r = 0; r < 8; ++r) {
                s[nf][r] *= scale;
                tmax[r] = fmaxf(tmax[r], s[nf][r]);
            }
#pragma unroll
        for (int r = 0; r < 8; ++r) {
            tmax[r] = fmaxf(tmax[r], __shfl_xor(tmax[r], 1, 32));
            tmax[r] = fmaxf(tmax[r], __shfl_xor(tmax[r], 2, 32));
            tmax[r] = fmaxf(tmax[r], __shfl_xor(tmax[r], 4, 32));
            tmax[r] = fmaxf(tmax[r], __shfl_xor(tmax[r], 8, 32));
            float mn = fmaxf(mrow[r], tmax[r]);
            scv[r]  = __expf(mrow[r] - mn);
            mrow[r] = mn;
            lrow[r] *= scv[r];
            rs[r] = 0.f;
        }
#pragma unroll
        for (int nf = 0; nf < 8; ++nf)
#pragma unroll
            for (int r = 0; r < 8; ++r) {
                float p = __expf(s[nf][r] - mrow[r]);
                s[nf][r] = p;
                rs[r] += p;
            }
#pragma unroll
        for (int r = 0; r < 8; ++r) {
            rs[r] += __shfl_xor(rs[r], 1, 32);
            rs[r] += __shfl_xor(rs[r], 2, 32);
            rs[r] += __shfl_xor(rs[r], 4, 32);
            rs[r] += __shfl_xor(rs[r], 8, 32);
            lrow[r] += rs[r];
        }
#pragma unroll
        for (int nf = 0; nf < 8; ++nf)
#pragma unroll
            for (int e = 0; e < 8; ++e) o[nf][e] *= scv[e];

        __syncthreads();                 // all waves done reading K tile
        // ---- P (C-layout) -> LDS [q][sk] for A-fragment reload ----
#pragma unroll
        for (int nf = 0; nf < 8; ++nf)
#pragma unroll
            for (int r = 0; r < 8; ++r)
                lds_kp[(wave * 16 + lh * 8 + r) * 128 + nf * 16 + l15] = (bf16)s[nf][r];
        __syncthreads();

        // ---- O += P * V ----
#pragma unroll
        for (int kf = 0; kf < 4; ++kf) {
            v16bf ap = ldsA_frag(lds_kp, wave * 16 + l15, kf * 32 + lh * 16, 128);
#pragma unroll
            for (int nf = 0; nf < 8; ++nf) {
                v16bf bv = ldsB_frag(lds_v, nf * 16 + l15, kf * 32 + lh * 16, 128);
                o[nf] = wmma_bf16(ap, bv, o[nf]);
            }
        }
    }

    // ---- normalize and store bf16 [token][qh*128+d] ----
#pragma unroll
    for (int nf = 0; nf < 8; ++nf)
#pragma unroll
        for (int r = 0; r < 8; ++r) {
            int row = qrow0 + wave * 16 + lh * 8 + r;
            int col = qh * HD + nf * 16 + l15;
            Oa[(size_t)row * DIM + col] = (bf16)(o[nf][r] / lrow[r]);
        }
}

// ---------------------------------------------------------------------------
extern "C" void kernel_launch(void* const* d_in, const int* in_sizes, int n_in,
                              void* d_out, int out_size, void* d_ws, size_t ws_size,
                              hipStream_t stream) {
    (void)in_sizes; (void)n_in; (void)out_size; (void)ws_size;

    const float* x    = (const float*)d_in[0];
    const float* wq_w = (const float*)d_in[1];
    const float* wq_b = (const float*)d_in[2];
    const float* wk_w = (const float*)d_in[3];
    const float* wk_b = (const float*)d_in[4];
    const float* wv_w = (const float*)d_in[5];
    const float* wv_b = (const float*)d_in[6];
    const float* wo_w = (const float*)d_in[7];
    const float* wo_b = (const float*)d_in[8];

    char* ws = (char*)d_ws;
    const size_t Q_BYTES = (size_t)TOKENS * DIM * sizeof(bf16);              // 32 MB
    const size_t K_BYTES = (size_t)TOKENS * KV_HEADS * HD * sizeof(bf16);    //  8 MB
    bf16* qws = (bf16*)(ws);
    bf16* kws = (bf16*)(ws + Q_BYTES);
    bf16* vws = (bf16*)(ws + Q_BYTES + K_BYTES);
    bf16* aws = (bf16*)(ws + Q_BYTES + 2 * K_BYTES);

    dim3 blk(256);
    // QKV projections (fp32 -> bf16 out)
    gemm_bias_wmma<false, true><<<dim3(DIM / 128, TOKENS / 128), blk, 0, stream>>>(
        x, wq_w, wq_b, qws, TOKENS, DIM, DIM);
    gemm_bias_wmma<false, true><<<dim3((KV_HEADS * HD) / 128, TOKENS / 128), blk, 0, stream>>>(
        x, wk_w, wk_b, kws, TOKENS, KV_HEADS * HD, DIM);
    gemm_bias_wmma<false, true><<<dim3((KV_HEADS * HD) / 128, TOKENS / 128), blk, 0, stream>>>(
        x, wv_w, wv_b, vws, TOKENS, KV_HEADS * HD, DIM);
    // attention: BATCH * Q_HEADS * (SLEN/128) blocks
    gqa_attention_kernel<<<dim3(BATCH * Q_HEADS * (SLEN / 128)), blk, 0, stream>>>(
        qws, kws, vws, aws);
    // output projection (bf16 A -> fp32 out)
    gemm_bias_wmma<true, false><<<dim3(DIM / 128, TOKENS / 128), blk, 0, stream>>>(
        aws, wo_w, wo_b, (float*)d_out, TOKENS, DIM, DIM);
}
